// Attention_38130719654007
// MI455X (gfx1250) — matrix operand order repaired and checked by reading, not compiled, not run on hardware
//
#include <hip/hip_runtime.h>

// ---------------- constants ----------------
constexpr int   BB    = 4;
constexpr int   TT    = 2305;
constexpr int   CC    = 384;
constexpr int   NH    = 6;
constexpr int   DH    = 64;
constexpr int   GS    = 48;          // 48x48 spatial grid
constexpr int   TP    = 2368;        // padded tokens per batch (148*16)
constexpr int   MROWS = BB * TT;     // 9220 token rows total
constexpr int   MT    = (MROWS + 15) / 16;  // 577 row tiles
constexpr int   NCH   = CC / 64;            // 6 column chunks of 64
constexpr float EPS   = 1e-5f;

typedef __attribute__((ext_vector_type(16))) __bf16       v16bf;
typedef __attribute__((ext_vector_type(8)))  float        v8f;
typedef __attribute__((ext_vector_type(4)))  unsigned int u32x4;
typedef __attribute__((ext_vector_type(4)))  int          i32x4;

union Frag16 {
  v16bf          v;
  u32x4          q[2];
  unsigned short h[16];
};

#if __has_builtin(__builtin_amdgcn_global_load_async_to_lds_b128)
#define HAVE_ASYNC_LDS 1
typedef __attribute__((address_space(1))) i32x4 gl_i32x4;  // global int4
typedef __attribute__((address_space(3))) i32x4 ld_i32x4;  // LDS int4
#else
#define HAVE_ASYNC_LDS 0
#endif

#define DEVFN static __device__ __forceinline__

DEVFN unsigned short f32_to_bf16(float f) {
  unsigned int u = __float_as_uint(f);
  u += 0x7FFFu + ((u >> 16) & 1u);        // round to nearest even
  return (unsigned short)(u >> 16);
}

// A fragment (16x32 bf16, MxK): lane L -> row m=L%16; halves 0..7 at k=hi*8..,
// halves 8..15 at k=16+hi*8.. (two contiguous 16B runs).
DEVFN v16bf load_a_frag(const unsigned short* base, int ld, int mBase, int kBase,
                        int lane, int mMax) {
  int m = mBase + (lane & 15);
  if (m >= mMax) m = mMax - 1;            // clamp: padded rows never stored
  int hi = lane >> 4;
  const unsigned short* p = base + (size_t)m * ld + kBase + hi * 8;
  Frag16 f;
  f.q[0] = *(const u32x4*)(p);
  f.q[1] = *(const u32x4*)(p + 16);
  return f.v;
}

// B fragment (32x16 bf16, KxN), element (k,n) read from base[n*ld + k]:
// lane L -> col n=L%16; halves 0..7 at k=hi*8.., halves 8..15 at k=16+hi*8..
DEVFN v16bf load_b_frag(const unsigned short* base, int ld, int kBase, int nBase,
                        int lane) {
  int n = nBase + (lane & 15);
  int hi = lane >> 4;
  const unsigned short* p = base + (size_t)n * ld + kBase + hi * 8;
  Frag16 f;
  f.q[0] = *(const u32x4*)(p);
  f.q[1] = *(const u32x4*)(p + 16);
  return f.v;
}

DEVFN v8f wmma_bf16(v16bf a, v16bf b, v8f c) {
  return __builtin_amdgcn_wmma_f32_16x16x32_bf16(false, a, false, b, (short)0, c,
                                                 false, false);
}

// ---------------- kernels ----------------

__global__ void zero_u32(unsigned int* p, size_t n) {
  size_t i = (size_t)blockIdx.x * blockDim.x + threadIdx.x;
  size_t st = (size_t)gridDim.x * blockDim.x;
  for (; i < n; i += st) p[i] = 0u;
}

__global__ void convert_weights(const float* wq, const float* wk, const float* wv,
                                const float* wo, unsigned short* oq,
                                unsigned short* ok, unsigned short* ov,
                                unsigned short* oo) {
  int i = blockIdx.x * blockDim.x + threadIdx.x;
  if (i < CC * CC) {
    oq[i] = f32_to_bf16(wq[i]);
    ok[i] = f32_to_bf16(wk[i]);
    ov[i] = f32_to_bf16(wv[i]);
    oo[i] = f32_to_bf16(wo[i]);
  }
}

// Fused depthwise 3x3 conv + BN for q/k/v; cls token (t==0) passed through.
__global__ __launch_bounds__(CC) void dwconv_bn_qkv(
    const float* __restrict__ x, const float* __restrict__ kq,
    const float* __restrict__ kk, const float* __restrict__ kv,
    const float* __restrict__ gq, const float* __restrict__ bq,
    const float* __restrict__ mq, const float* __restrict__ vq,
    const float* __restrict__ gk, const float* __restrict__ bk,
    const float* __restrict__ mk, const float* __restrict__ vk,
    const float* __restrict__ gv, const float* __restrict__ bv,
    const float* __restrict__ mv, const float* __restrict__ vv,
    unsigned short* __restrict__ qin, unsigned short* __restrict__ kin,
    unsigned short* __restrict__ vin) {
  int t = blockIdx.x, b = blockIdx.y, c = threadIdx.x;
  size_t o = ((size_t)b * TT + t) * CC + c;
  if (t == 0) {
    unsigned short cls = f32_to_bf16(x[o]);
    qin[o] = cls; kin[o] = cls; vin[o] = cls;
    return;
  }
  int s = t - 1, yy = s / GS, xx = s - yy * GS;
  float aq = 0.f, ak = 0.f, av = 0.f;
  #pragma unroll
  for (int dy = -1; dy <= 1; ++dy) {
    int y2 = yy + dy;
    if ((unsigned)y2 >= (unsigned)GS) continue;
    #pragma unroll
    for (int dx = -1; dx <= 1; ++dx) {
      int x2 = xx + dx;
      if ((unsigned)x2 >= (unsigned)GS) continue;
      float xv = x[((size_t)b * TT + 1 + y2 * GS + x2) * CC + c];
      int ki = c * 9 + (dy + 1) * 3 + (dx + 1);
      aq = fmaf(xv, kq[ki], aq);
      ak = fmaf(xv, kk[ki], ak);
      av = fmaf(xv, kv[ki], av);
    }
  }
  aq = (aq - mq[c]) * (gq[c] * rsqrtf(vq[c] + EPS)) + bq[c];
  ak = (ak - mk[c]) * (gk[c] * rsqrtf(vk[c] + EPS)) + bk[c];
  av = (av - mv[c]) * (gv[c] * rsqrtf(vv[c] + EPS)) + bv[c];
  qin[o] = f32_to_bf16(aq);
  kin[o] = f32_to_bf16(ak);
  vin[o] = f32_to_bf16(av);
}

// out[t,n] = sum_c in[t,c] * W[n,c] ; 16x64 strip per wave (one head chunk),
// scatter into padded (B,H,TP,DH) head buffer.
__global__ __launch_bounds__(128) void gemm_qkv(
    const unsigned short* __restrict__ in, const unsigned short* __restrict__ w,
    unsigned short* __restrict__ outb) {
  int lane = threadIdx.x & 31, wid = threadIdx.x >> 5;
  int tile = blockIdx.x * 4 + wid;
  if (tile >= MT * NCH) return;
  int mBase = (tile / NCH) * 16, nBase = (tile % NCH) * 64;
  v8f acc[4] = {{}, {}, {}, {}};
  #pragma unroll 2
  for (int k = 0; k < CC; k += 32) {
    v16bf a = load_a_frag(in, CC, mBase, k, lane, MROWS);
    #pragma unroll
    for (int j = 0; j < 4; ++j)
      acc[j] = wmma_bf16(a, load_b_frag(w, CC, k, nBase + 16 * j, lane), acc[j]);
  }
  int nl = lane & 15, hi = lane >> 4;
  int head = nBase / DH;                 // wave-uniform: 64-chunk == one head
  #pragma unroll
  for (int r = 0; r < 8; ++r) {
    int m = mBase + r + 8 * hi;
    if (m < MROWS) {
      int b = m / TT, t = m - b * TT;
      size_t rowo = ((size_t)(b * NH + head) * TP + t) * DH;
      #pragma unroll
      for (int j = 0; j < 4; ++j)
        outb[rowo + 16 * j + nl] = f32_to_bf16(acc[j][r]);
    }
  }
}

// Flash attention: one wave per 16-row Q strip, 4 waves/block sharing K/V tiles.
__global__ __launch_bounds__(128) void flash_attn(
    const unsigned short* __restrict__ qb, const unsigned short* __restrict__ kb,
    const unsigned short* __restrict__ vb, unsigned short* __restrict__ ao) {
  __shared__ __align__(16) unsigned short Kt[32 * 64];   // [key][d]
  __shared__ __align__(16) unsigned short Vt[64 * 32];   // transposed [d][key]
  __shared__ __align__(16) unsigned short Ps[4][16 * 32];

  int lane = threadIdx.x & 31, wid = threadIdx.x >> 5;
  int h = blockIdx.y, b = blockIdx.z;
  int qBase = (blockIdx.x * 4 + wid) * 16;   // < TP, all tiles valid
  const unsigned short* Q = qb + (size_t)(b * NH + h) * TP * DH;
  const unsigned short* K = kb + (size_t)(b * NH + h) * TP * DH;
  const unsigned short* V = vb + (size_t)(b * NH + h) * TP * DH;

  v16bf qf0 = load_a_frag(Q, DH, qBase, 0, lane, TP);
  v16bf qf1 = load_a_frag(Q, DH, qBase, 32, lane, TP);

  v8f o0 = {}, o1 = {}, o2 = {}, o3 = {};
  float mrow[8], lrow[8];
  #pragma unroll
  for (int r = 0; r < 8; ++r) { mrow[r] = -1e30f; lrow[r] = 0.f; }

  const float scale = rsqrtf((float)CC);   // reference: c**-0.5
  const int nsteps = (TT + 31) / 32;       // 73; covers keys < 2336 <= TP

  int srow = threadIdx.x >> 2;             // 0..31 staging key row
  int scol = (threadIdx.x & 3) * 16;       // staging d chunk

  for (int step = 0; step < nsteps; ++step) {
    int ktBase = step * 32;
    __syncthreads();
    {  // cooperative stage: K row-major (async if available), V transposed
      const unsigned short* gk = K + (size_t)(ktBase + srow) * DH + scol;
#if HAVE_ASYNC_LDS
      __builtin_amdgcn_global_load_async_to_lds_b128(
          (gl_i32x4*)gk, (ld_i32x4*)&Kt[srow * 64 + scol], 0, 0);
      __builtin_amdgcn_global_load_async_to_lds_b128(
          (gl_i32x4*)(gk + 8), (ld_i32x4*)&Kt[srow * 64 + scol + 8], 0, 0);
#else
      *(u32x4*)(&Kt[srow * 64 + scol])     = *(const u32x4*)(gk);
      *(u32x4*)(&Kt[srow * 64 + scol + 8]) = *(const u32x4*)(gk + 8);
#endif
      const unsigned short* gv = V + (size_t)(ktBase + srow) * DH + scol;
      #pragma unroll
      for (int i = 0; i < 16; ++i) Vt[(scol + i) * 32 + srow] = gv[i];
      if (step + 1 < nsteps) {  // overlap next tile's HBM latency (prefetch)
        __builtin_prefetch(gk + 32 * DH, 0, 0);
        __builtin_prefetch(gv + 32 * DH, 0, 0);
      }
#if HAVE_ASYNC_LDS
      asm volatile("s_wait_asynccnt 0" ::: "memory");
#endif
    }
    __syncthreads();

    // S = Q K^T : two 16-key subtiles, K-dim = DH in two 32-chunks
    v8f s0 = {}, s1 = {};
    #pragma unroll
    for (int dc = 0; dc < 2; ++dc) {
      v16bf qa = dc ? qf1 : qf0;
      s0 = wmma_bf16(qa, load_b_frag(Kt, 64, dc * 32, 0, lane), s0);
      s1 = wmma_bf16(qa, load_b_frag(Kt, 64, dc * 32, 16, lane), s1);
    }

    int hi = lane >> 4, nl = lane & 15;
    int kt0 = ktBase + nl, kt1 = ktBase + 16 + nl;
    #pragma unroll
    for (int r = 0; r < 8; ++r) {
      float e0 = (kt0 < TT) ? s0[r] * scale : -1e30f;
      float e1 = (kt1 < TT) ? s1[r] * scale : -1e30f;
      float vmx = fmaxf(e0, e1);
      #pragma unroll
      for (int off = 8; off >= 1; off >>= 1)
        vmx = fmaxf(vmx, __shfl_xor(vmx, off, 32));  // stays in 16-lane half
      float mnew = fmaxf(mrow[r], vmx);
      float corr = __expf(mrow[r] - mnew);
      float p0 = __expf(e0 - mnew);
      float p1 = __expf(e1 - mnew);
      float rs = p0 + p1;
      #pragma unroll
      for (int off = 8; off >= 1; off >>= 1) rs += __shfl_xor(rs, off, 32);
      lrow[r] = lrow[r] * corr + rs;
      mrow[r] = mnew;
      o0[r] *= corr; o1[r] *= corr; o2[r] *= corr; o3[r] *= corr;
      int prow = r + 8 * hi;                       // C/D layout row for this lane
      Ps[wid][prow * 32 + nl]      = f32_to_bf16(p0);
      Ps[wid][prow * 32 + 16 + nl] = f32_to_bf16(p1);
    }
    asm volatile("s_wait_dscnt 0" ::: "memory");   // wave-local LDS transpose fence

    v16bf pa = load_a_frag(Ps[wid], 32, 0, 0, lane, 16);
    o0 = wmma_bf16(pa, load_b_frag(Vt, 32, 0, 0,  lane), o0);
    o1 = wmma_bf16(pa, load_b_frag(Vt, 32, 0, 16, lane), o1);
    o2 = wmma_bf16(pa, load_b_frag(Vt, 32, 0, 32, lane), o2);
    o3 = wmma_bf16(pa, load_b_frag(Vt, 32, 0, 48, lane), o3);
  }

  int hi = lane >> 4, nl = lane & 15;
  #pragma unroll
  for (int r = 0; r < 8; ++r) {
    int trow = qBase + r + 8 * hi;
    if (trow < TT) {
      float inv = 1.f / lrow[r];
      size_t rowo = ((size_t)b * TT + trow) * CC + h * DH;
      ao[rowo + 0  + nl] = f32_to_bf16(o0[r] * inv);
      ao[rowo + 16 + nl] = f32_to_bf16(o1[r] * inv);
      ao[rowo + 32 + nl] = f32_to_bf16(o2[r] * inv);
      ao[rowo + 48 + nl] = f32_to_bf16(o3[r] * inv);
    }
  }
}

// y = ao @ Wo^T + bo, 16x64 strip per wave, f32 output
__global__ __launch_bounds__(128) void gemm_out(
    const unsigned short* __restrict__ in, const unsigned short* __restrict__ w,
    const float* __restrict__ bo, float* __restrict__ out) {
  int lane = threadIdx.x & 31, wid = threadIdx.x >> 5;
  int tile = blockIdx.x * 4 + wid;
  if (tile >= MT * NCH) return;
  int mBase = (tile / NCH) * 16, nBase = (tile % NCH) * 64;
  v8f acc[4] = {{}, {}, {}, {}};
  #pragma unroll 2
  for (int k = 0; k < CC; k += 32) {
    v16bf a = load_a_frag(in, CC, mBase, k, lane, MROWS);
    #pragma unroll
    for (int j = 0; j < 4; ++j)
      acc[j] = wmma_bf16(a, load_b_frag(w, CC, k, nBase + 16 * j, lane), acc[j]);
  }
  int nl = lane & 15, hi = lane >> 4;
  float bias[4];
  #pragma unroll
  for (int j = 0; j < 4; ++j) bias[j] = bo[nBase + 16 * j + nl];
  #pragma unroll
  for (int r = 0; r < 8; ++r) {
    int m = mBase + r + 8 * hi;
    if (m < MROWS) {
      #pragma unroll
      for (int j = 0; j < 4; ++j)
        out[(size_t)m * CC + nBase + 16 * j + nl] = acc[j][r] + bias[j];
    }
  }
}

// ---------------- launch ----------------
extern "C" void kernel_launch(void* const* d_in, const int* in_sizes, int n_in,
                              void* d_out, int out_size, void* d_ws,
                              size_t ws_size, hipStream_t stream) {
  const float* x  = (const float*)d_in[0];
  const float* kq = (const float*)d_in[3];
  const float* kk = (const float*)d_in[4];
  const float* kv = (const float*)d_in[5];
  const float* gq = (const float*)d_in[6];
  const float* bq = (const float*)d_in[7];
  const float* mq = (const float*)d_in[8];
  const float* vq = (const float*)d_in[9];
  const float* gk = (const float*)d_in[10];
  const float* bk = (const float*)d_in[11];
  const float* mk = (const float*)d_in[12];
  const float* vk = (const float*)d_in[13];
  const float* gv = (const float*)d_in[14];
  const float* bv = (const float*)d_in[15];
  const float* mv = (const float*)d_in[16];
  const float* vv = (const float*)d_in[17];
  const float* Wq = (const float*)d_in[18];
  const float* Wk = (const float*)d_in[19];
  const float* Wv = (const float*)d_in[20];
  const float* Wo = (const float*)d_in[21];
  const float* bo = (const float*)d_in[22];
  float* out = (float*)d_out;

  char* ws = (char*)d_ws;
  size_t off = 0;
  auto alloc = [&](size_t bytes) {
    size_t o = off;
    off = (off + bytes + 255) & ~(size_t)255;
    return o;
  };
  const size_t WB   = (size_t)CC * CC * sizeof(unsigned short);
  const size_t INB  = (size_t)MROWS * CC * sizeof(unsigned short);
  const size_t QKVN = (size_t)BB * NH * TP * DH;  // elements per head buffer

  unsigned short* wqb = (unsigned short*)(ws + alloc(WB));
  unsigned short* wkb = (unsigned short*)(ws + alloc(WB));
  unsigned short* wvb = (unsigned short*)(ws + alloc(WB));
  unsigned short* wob = (unsigned short*)(ws + alloc(WB));
  unsigned short* qin = (unsigned short*)(ws + alloc(INB));
  unsigned short* kin = (unsigned short*)(ws + alloc(INB));
  unsigned short* vin = (unsigned short*)(ws + alloc(INB));
  unsigned short* qb  = (unsigned short*)(ws + alloc(QKVN * 2));
  unsigned short* kb  = (unsigned short*)(ws + alloc(QKVN * 2));
  unsigned short* vb  = (unsigned short*)(ws + alloc(QKVN * 2));
  unsigned short* ao  = (unsigned short*)(ws + alloc(INB));

  // 1. weights -> bf16
  convert_weights<<<(CC * CC + 255) / 256, 256, 0, stream>>>(Wq, Wk, Wv, Wo, wqb,
                                                             wkb, wvb, wob);
  // 2. zero the padded head buffers (qb/kb/vb are contiguous)
  zero_u32<<<1024, 256, 0, stream>>>((unsigned int*)qb, 3 * QKVN / 2);
  // 3. depthwise conv + BN
  dwconv_bn_qkv<<<dim3(TT, BB), CC, 0, stream>>>(x, kq, kk, kv, gq, bq, mq, vq,
                                                 gk, bk, mk, vk, gv, bv, mv, vv,
                                                 qin, kin, vin);
  // 4. QKV projections (WMMA), 16x64 strip per wave
  int gblocks = (MT * NCH + 3) / 4;  // 3462 wave-strips / 4
  gemm_qkv<<<gblocks, 128, 0, stream>>>(qin, wqb, qb);
  gemm_qkv<<<gblocks, 128, 0, stream>>>(kin, wkb, kb);
  gemm_qkv<<<gblocks, 128, 0, stream>>>(vin, wvb, vb);
  // 5. flash attention (WMMA) : 148 q-tiles / 4 waves = 37 blocks per (b,h)
  flash_attn<<<dim3(TP / 64, NH, BB), 128, 0, stream>>>(qb, kb, vb, ao);
  // 6. output projection + bias (WMMA, f32 out)
  gemm_out<<<gblocks, 128, 0, stream>>>(ao, wob, bo, out);
}
